// ctc_hinge_loss_57896159150511
// MI455X (gfx1250) — matrix operand order repaired and checked by reading, not compiled, not run on hardware
//
#include <hip/hip_runtime.h>
#include <hip/hip_bf16.h>
#include <stdint.h>

// ---------------------------------------------------------------------------
// CTC "hinge" loss for MI455X (gfx1250, wave32).
// out = (1 - L) + L  with L = sum_b CTC_NLL(b).
//
// kernel 1: lse_kernel        — per-(t,b) logsumexp over V (wave-per-row).
// kernel 2: ctc_scan_kernel   — 1 wave32 per batch element, barrier-free scan:
//            * async global->LDS b128 streaming of logit rows (ASYNCcnt),
//              3-chunk (48-step) software pipeline,
//            * per-16-step emission tile E = Acts(f16) x OneHot(f16) via
//              v_wmma_f32_16x16x32_f16; one-hot fragments precomputed once
//              into LDS (loop-invariant), reloaded as 2x ds_load_b128,
//            * alphas in registers, s-1/s-2 neighbors via __shfl_up.
// kernel 3: finalize_kernel   — deterministic tree sum, out=(1-L)+L.
// ---------------------------------------------------------------------------

typedef _Float16 v16h __attribute__((ext_vector_type(16)));
typedef _Float16 v8h  __attribute__((ext_vector_type(8)));
typedef float    v8f  __attribute__((ext_vector_type(8)));

#define T_   2000
#define B_   128
#define V_   128
#define L_   100
#define S_   201      // 2L+1
#define SP_  224      // padded state count = 32 lanes * 7 states
#define NCH_ 125      // T_/16 chunks of 16 timesteps
#define NEGF (-1e30f)

// ---- CDNA5 async global->LDS copy (tracked by ASYNCcnt) --------------------
__device__ __forceinline__ void async_ld_b128(uint32_t lds_off, uint32_t g_off,
                                              const void* sbase) {
  asm volatile("global_load_async_to_lds_b128 %0, %1, %2"
               :: "v"(lds_off), "v"(g_off), "s"(sbase) : "memory");
}
template <int N>
__device__ __forceinline__ void wait_asynccnt() {
  asm volatile("s_wait_asynccnt %0" :: "i"(N) : "memory");
}
__device__ __forceinline__ void wait_dscnt0() {
  asm volatile("s_wait_dscnt 0" ::: "memory");
}

// ---- log(exp(x)+exp(y)+exp(z)) robust at NEGF (max3 -> v_max3_f32) ---------
__device__ __forceinline__ float lae3(float x, float y, float z) {
  float m = fmaxf(x, fmaxf(y, z));
  return m + __logf(__expf(x - m) + __expf(y - m) + __expf(z - m));
}

// ===========================================================================
// Kernel 1: lse[t*B+b] = logsumexp_v acts[t,b,v].  One wave32 per row.
// ===========================================================================
__global__ __launch_bounds__(256) void lse_kernel(const float* __restrict__ acts,
                                                  float* __restrict__ lse) {
  const int row  = blockIdx.x * 8 + (threadIdx.x >> 5);   // 8 rows per block
  const int lane = threadIdx.x & 31;
  const float4 v = *(const float4*)(acts + (size_t)row * V_ + lane * 4);
  float m = fmaxf(fmaxf(v.x, v.y), fmaxf(v.z, v.w));
#pragma unroll
  for (int o = 16; o; o >>= 1) m = fmaxf(m, __shfl_xor(m, o, 32));
  float s = __expf(v.x - m) + __expf(v.y - m) + __expf(v.z - m) + __expf(v.w - m);
#pragma unroll
  for (int o = 16; o; o >>= 1) s += __shfl_xor(s, o, 32);
  if (lane == 0) lse[row] = m + __logf(s);
}

// ===========================================================================
// Kernel 2: CTC forward scan, one wave32 per batch element.
// ===========================================================================
__global__ __launch_bounds__(32) void ctc_scan_kernel(
    const float* __restrict__ acts, const int* __restrict__ labels,
    const int* __restrict__ act_lens, const int* __restrict__ label_lens,
    const float* __restrict__ lse, float* __restrict__ loss_out) {
  const int b     = blockIdx.x;
  const int lane  = threadIdx.x;
  const int lhalf = lane >> 4;     // 0: lanes 0-15, 1: lanes 16-31
  const int lm    = lane & 15;

  __shared__ float    ring[3][16 * V_];     // 3-chunk async landing buffer (24KB)
  __shared__ _Float16 tile[16 * V_];        // f16 acts chunk, time-major (4KB)
  __shared__ float    Ebuf[2][16 * SP_];    // emission tiles, double buffered (28KB)
  __shared__ _Float16 bfl[52][32][16];      // precomputed one-hot B frags (53KB)
  __shared__ int      ext[SP_];
  __shared__ float    skipadd[SP_];         // 0 if skip transition allowed else NEGF
  __shared__ float    aout[SP_];

  // --- extended label sequence + skip mask ---------------------------------
  for (int s = lane; s < SP_; s += 32)
    ext[s] = ((s & 1) && s < S_) ? labels[b * L_ + (s >> 1)] : 0;
  __syncthreads();
  for (int s = lane; s < SP_; s += 32) {
    bool cs = (s >= 2) && (s < S_) && (ext[s] != 0) && (ext[s] != ext[s - 2]);
    skipadd[s] = cs ? 0.0f : NEGF;
  }
  __syncthreads();

  float skr[7];
#pragma unroll
  for (int j = 0; j < 7; j++) skr[j] = skipadd[lane * 7 + j];

  // --- one-hot WMMA B fragments: loop-invariant, built once into LDS -------
  // B frag (st,kk) is 32x16 f16: lanes 0-15 hold K 0-15 of slice kk (column
  // n = lane&15), lanes 16-31 hold K 16-31.  bfl[f][lane][i] is half i of
  // lane's fragment; reloaded per chunk as two ds_load_b128.
#pragma unroll
  for (int st = 0; st < 13; st++) {
    const int extv = ext[st * 16 + lm];
#pragma unroll
    for (int kk = 0; kk < 4; kk++) {
      const int delta = extv - (kk * 32 + lhalf * 16);
#pragma unroll
      for (int i = 0; i < 16; i++)
        bfl[st * 4 + kk][lane][i] = (_Float16)((delta == i) ? 1.0f : 0.0f);
    }
  }
  __syncthreads();

  const int Tb  = act_lens[b];
  const int ll  = label_lens[b];
  int nch = (Tb + 15) >> 4; if (nch > NCH_) nch = NCH_;

  float a[7];
#pragma unroll
  for (int j = 0; j < 7; j++) a[j] = NEGF;

  // --- pipeline helpers ----------------------------------------------------
  auto issue_chunk = [&](int c) {
    if (c >= NCH_) return;
    const uint32_t ldsbase =
        (uint32_t)(uintptr_t)(&ring[c % 3][0]) + (uint32_t)(lane * 16);
#pragma unroll
    for (int r = 0; r < 16; r++) {
      uint32_t goff = (uint32_t)(((c * 16 + r) * B_ + b) * (V_ * 4)) +
                      (uint32_t)(lane * 16);
      async_ld_b128(ldsbase + (uint32_t)(r * V_ * 4), goff, acts);
    }
  };

  auto convert = [&](int c) {            // f32 ring chunk -> f16 tile
    const float* sp = &ring[c % 3][0];
    for (int i = lane * 2; i < 16 * V_; i += 64) {
      tile[i]     = (_Float16)sp[i];
      tile[i + 1] = (_Float16)sp[i + 1];
    }
  };

  auto buildE = [&](int c) {             // E[tt][s] = acts[t0+tt, ext[s]] via WMMA
    const int par = c & 1;
    // A fragments: 16x32 f16 acts (M=tt, K=vocab slice). contiguous k order (same as B):
    // lanes 0-15 hold K 0-15, lanes 16-31 hold K 16-31 (matches B frags).
    v16h afr[4];
#pragma unroll
    for (int kk = 0; kk < 4; kk++) {
      const _Float16* base = &tile[lm * V_ + kk * 32 + lhalf * 16];
      v8h lo = *(const v8h*)(base);
      v8h hi = *(const v8h*)(base + 8);
      v16h av;
#pragma unroll
      for (int i = 0; i < 8; i++) { av[i] = lo[i]; av[i + 8] = hi[i]; }
      afr[kk] = av;
    }
    float* ep = &Ebuf[par][0];
#pragma unroll
    for (int st = 0; st < 13; st++) {
      v8f acc = {};
#pragma unroll
      for (int kk = 0; kk < 4; kk++) {
        const _Float16* bp = &bfl[st * 4 + kk][lane][0];
        v8h lo = *(const v8h*)(bp);
        v8h hi = *(const v8h*)(bp + 8);
        v16h bf;
#pragma unroll
        for (int i = 0; i < 8; i++) { bf[i] = lo[i]; bf[i + 8] = hi[i]; }
        acc = __builtin_amdgcn_wmma_f32_16x16x32_f16(
            false, afr[kk], false, bf, (short)0, acc, false, false);
      }
      // C layout: VGPR j -> (M = j + 8*lhalf, N = lm)
#pragma unroll
      for (int j = 0; j < 8; j++)
        ep[(j + 8 * lhalf) * SP_ + st * 16 + lm] = acc[j];
    }
  };

  auto scan_chunk = [&](int c) {
    const float* ep = &Ebuf[c & 1][0];
    const float  lsev = lse[(c * 16 + lm) * B_ + b];   // one load, bcast by shfl
#pragma unroll
    for (int tt = 0; tt < 16; tt++) {
      const int t = c * 16 + tt;
      if (t >= Tb) break;                               // wave-uniform freeze
      const float lt = __shfl(lsev, tt, 32);
      const float* er = ep + tt * SP_ + lane * 7;
      const float e0 = er[0] - lt, e1 = er[1] - lt, e2 = er[2] - lt,
                  e3 = er[3] - lt, e4 = er[4] - lt, e5 = er[5] - lt,
                  e6 = er[6] - lt;
      if (t == 0) {
        a[0] = (lane == 0) ? e0 : NEGF;
        a[1] = (lane == 0) ? e1 : NEGF;
        a[2] = a[3] = a[4] = a[5] = a[6] = NEGF;
      } else {
        float p1 = __shfl_up(a[6], 1, 32);
        float p2 = __shfl_up(a[5], 1, 32);
        p1 = lane ? p1 : NEGF;
        p2 = lane ? p2 : NEGF;
        const float o0 = a[0], o1 = a[1], o2 = a[2], o3 = a[3], o4 = a[4],
                    o5 = a[5], o6 = a[6];
        a[0] = lae3(o0, p1, p2 + skr[0]) + e0;
        a[1] = lae3(o1, o0, p1 + skr[1]) + e1;
        a[2] = lae3(o2, o1, o0 + skr[2]) + e2;
        a[3] = lae3(o3, o2, o1 + skr[3]) + e3;
        a[4] = lae3(o4, o3, o2 + skr[4]) + e4;
        a[5] = lae3(o5, o4, o3 + skr[5]) + e5;
        a[6] = lae3(o6, o5, o4 + skr[6]) + e6;
      }
    }
  };

  // --- software pipeline: build(c) overlaps scan(c-1); single inline site --
  issue_chunk(0); issue_chunk(1); issue_chunk(2);
  for (int c = 0; c < nch; ++c) {
    wait_asynccnt<32>();        // chunk c resident; c+1,c+2 still in flight
    convert(c);
    wait_dscnt0();              // ring slot reads done before re-issuing into it
    issue_chunk(c + 3);         // refill the slot just drained
    buildE(c);
    if (c) scan_chunk(c - 1);   // wave-uniform branch
  }
  scan_chunk(nch - 1);

  // --- extract log-likelihood ---------------------------------------------
#pragma unroll
  for (int j = 0; j < 7; j++) aout[lane * 7 + j] = a[j];
  __syncthreads();
  if (lane == 0) {
    const int eb = 2 * ll;
    const int el = (2 * ll - 1 > 0) ? (2 * ll - 1) : 0;
    const float ab = aout[eb], al = aout[el];
    const float m = fmaxf(ab, al);
    const float lik = m + __logf(__expf(ab - m) + __expf(al - m));
    loss_out[b] = -lik;
  }
}

// ===========================================================================
// Kernel 3: deterministic sum of per-batch losses; out = (1 - L) + L.
// ===========================================================================
__global__ __launch_bounds__(128) void finalize_kernel(const float* __restrict__ loss,
                                                       float* __restrict__ out) {
  __shared__ float sm[128];
  sm[threadIdx.x] = loss[threadIdx.x];
  __syncthreads();
#pragma unroll
  for (int s = 64; s; s >>= 1) {
    if ((int)threadIdx.x < s) sm[threadIdx.x] += sm[threadIdx.x + s];
    __syncthreads();
  }
  if (threadIdx.x == 0) {
    const float L = sm[0];
    out[0] = (1.0f - L) + L;   // matches aug_loss - l1 + l2 with l1 == l2
  }
}

// ===========================================================================
extern "C" void kernel_launch(void* const* d_in, const int* in_sizes, int n_in,
                              void* d_out, int out_size, void* d_ws, size_t ws_size,
                              hipStream_t stream) {
  const float* acts       = (const float*)d_in[0];
  const int*   labels     = (const int*)d_in[1];
  const int*   act_lens   = (const int*)d_in[2];
  const int*   label_lens = (const int*)d_in[3];
  float*       out        = (float*)d_out;

  float* ws_lse  = (float*)d_ws;                 // T*B floats (1.0 MB)
  float* ws_loss = ws_lse + (size_t)T_ * B_;     // B floats

  lse_kernel<<<(T_ * B_) / 8, 256, 0, stream>>>(acts, ws_lse);
  ctc_scan_kernel<<<B_, 32, 0, stream>>>(acts, labels, act_lens, label_lens,
                                         ws_lse, ws_loss);
  finalize_kernel<<<1, 128, 0, stream>>>(ws_loss, out);
}
